// SphericalTransformer_53188874994217
// MI455X (gfx1250) — matrix operand order repaired and checked by reading, not compiled, not run on hardware
//
#include <hip/hip_runtime.h>
#include <hip/hip_bf16.h>
#include <math.h>

typedef unsigned short u16;
typedef __bf16 bf16_t;
typedef bf16_t v8bf  __attribute__((ext_vector_type(8)));
typedef bf16_t v16bf __attribute__((ext_vector_type(16)));
typedef float  v8f   __attribute__((ext_vector_type(8)));

#define NTOK   2048          // 32*64 tokens per image
#define MROWS  4096          // B * NTOK
#define DIMC   256
#define NHEADS 8
#define HDIM   32
#define HIDDENC 1024

// ---------------------------------------------------------------- helpers
__device__ __forceinline__ u16 f2bf(float f){
  unsigned u = __float_as_uint(f);
  u += 0x7fffu + ((u >> 16) & 1u);          // round-to-nearest-even
  return (u16)(u >> 16);
}
__device__ __forceinline__ v16bf cat8(v8bf lo, v8bf hi){
  return __builtin_shufflevector(lo, hi, 0,1,2,3,4,5,6,7,8,9,10,11,12,13,14,15);
}
__device__ __forceinline__ v8f wmma_bf16(v16bf a, v16bf b, v8f c){
  // D = A(16x32) * B(32x16) + C, f32 accumulate
  return __builtin_amdgcn_wmma_f32_16x16x32_bf16(false, a, false, b, (short)0, c, false, false);
}
// A fragment: lane<16 holds row (lane&15), K = k0+0..7 then k0+16..23;
// lane>=16 holds same row, K = k0+8..15 then k0+24..31 (ISA 7.12.2).
__device__ __forceinline__ v16bf load_a_frag(const u16* A, int lda, int m0, int k0, int lane){
  int row  = m0 + (lane & 15);
  int half = lane >> 4;
  const u16* p = A + (size_t)row * lda + k0 + 8 * half;
  v8bf lo = *(const v8bf*)p;
  v8bf hi = *(const v8bf*)(p + 16);
  return cat8(lo, hi);
}
// B fragment: lane holds column n0+(lane&15); K = k0+8*(lane>>4)+0..7 then +16..23.
// W stored row-major (N x K) == column-major B, so rows of W load contiguously.
__device__ __forceinline__ v16bf load_b_frag(const u16* W, int ldw, int n0, int k0, int lane){
  int col  = n0 + (lane & 15);
  int half = lane >> 4;
  const u16* p = W + (size_t)col * ldw + k0 + 8 * half;
  v8bf lo = *(const v8bf*)p;
  v8bf hi = *(const v8bf*)(p + 16);
  return cat8(lo, hi);
}
__device__ __forceinline__ float gelu_exact(float x){
  return 0.5f * x * (1.f + erff(x * 0.70710678118654752f));
}

// ---------------------------------------------------------------- tiny prep kernels
__global__ void k_f2bf(const float* __restrict__ s, u16* __restrict__ d, int n){
  int i = blockIdx.x * blockDim.x + threadIdx.x;
  if(i < n) d[i] = f2bf(s[i]);
}
__global__ void k_bnfuse(const float* g, const float* b, const float* m, const float* v,
                         float* sc, float* sh, int n){
  int i = blockIdx.x * blockDim.x + threadIdx.x;
  if(i < n){
    float inv = rsqrtf(v[i] + 1e-5f);
    sc[i] = g[i] * inv;
    sh[i] = b[i] - m[i] * g[i] * inv;
  }
}
// im2col for conv1: x(2,3,256,512) -> A1 (4096 x 192), k = ci*64 + dy*8 + dx (OIHW order)
__global__ void k_im2col1(const float* __restrict__ x, u16* __restrict__ A1){
  int idx = blockIdx.x * blockDim.x + threadIdx.x;
  if(idx >= MROWS * 192) return;
  int r = idx / 192, k = idx - r * 192;
  int ci = k >> 6, rem = k & 63, dy = rem >> 3, dx = rem & 7;
  int b = r >> 11, t = r & 2047, py = t >> 6, px = t & 63;
  float v = x[(((size_t)b * 3 + ci) * 256 + py * 8 + dy) * 512 + px * 8 + dx];
  A1[idx] = f2bf(v);
}
// im2col for conv2 (3x3, pad 1) over y1b (rows=(b,py,px), 128ch) -> A2 (4096 x 1152)
__global__ void k_im2col2(const u16* __restrict__ y1, u16* __restrict__ A2){
  int idx = blockIdx.x * blockDim.x + threadIdx.x;
  if(idx >= MROWS * 1152) return;
  int r = idx / 1152, k = idx - r * 1152;
  int ci = k / 9, rem = k - ci * 9, ky = rem / 3, kx = rem - ky * 3;
  int b = r >> 11, t = r & 2047, py = t >> 6, px = t & 63;
  int yy = py + ky - 1, xx = px + kx - 1;
  u16 v = 0;
  if(yy >= 0 && yy < 32 && xx >= 0 && xx < 64)
    v = y1[((size_t)(b * 2048 + yy * 64 + xx)) * 128 + ci];
  A2[idx] = v;
}

// ---------------------------------------------------------------- WMMA GEMM
// C(M=4096 x N) = act( A(bf16, MxK row-major) * W(bf16, NxK row-major)^T * scale + shift )
// mode bit0: outF = v ; bit1: outF += v ; bit2: outB = bf16(v)
__global__ __launch_bounds__(256) void k_gemm(
    const u16* __restrict__ A, const u16* __restrict__ W, int K, int N,
    const float* __restrict__ scale, const float* __restrict__ shift,
    int act, int mode, float* __restrict__ outF, u16* __restrict__ outB)
{
  int lane = threadIdx.x & 31, w = threadIdx.x >> 5;
  int m0 = blockIdx.x * 128 + w * 16;       // 8 waves stacked in M
  int n0 = blockIdx.y * 64;                 // each wave covers 64 columns
  v8f c0, c1, c2, c3;
  #pragma unroll
  for(int r = 0; r < 8; ++r){ c0[r] = 0.f; c1[r] = 0.f; c2[r] = 0.f; c3[r] = 0.f; }
  for(int k0 = 0; k0 < K; k0 += 32){
    v16bf a  = load_a_frag(A, K, m0, k0, lane);            // reused 4x
    v16bf b0 = load_b_frag(W, K, n0,      k0, lane);
    v16bf b1 = load_b_frag(W, K, n0 + 16, k0, lane);
    v16bf b2 = load_b_frag(W, K, n0 + 32, k0, lane);
    v16bf b3 = load_b_frag(W, K, n0 + 48, k0, lane);
    c0 = wmma_bf16(a, b0, c0);
    c1 = wmma_bf16(a, b1, c1);
    c2 = wmma_bf16(a, b2, c2);
    c3 = wmma_bf16(a, b3, c3);
  }
  int half = lane >> 4, ln = lane & 15;
  v8f acc[4] = {c0, c1, c2, c3};
  #pragma unroll
  for(int t = 0; t < 4; ++t){
    int col = n0 + 16 * t + ln;
    float sc = scale ? scale[col] : 1.f;
    float sh = shift ? shift[col] : 0.f;
    #pragma unroll
    for(int r = 0; r < 8; ++r){
      int row = m0 + r + 8 * half;          // C/D layout: VGPR r -> M = r (+8 for hi half)
      float v = acc[t][r] * sc + sh;
      if(act == 1) v = gelu_exact(v);
      size_t idx = (size_t)row * N + col;
      if(mode & 1) outF[idx] = v;
      if(mode & 2) outF[idx] += v;
      if(mode & 4) outB[idx] = f2bf(v);
    }
  }
}

// ---------------------------------------------------------------- LayerNorm (wave per row)
__global__ __launch_bounds__(256) void k_ln(const float* __restrict__ X,
    const float* __restrict__ g, const float* __restrict__ b,
    u16* __restrict__ outB, float* __restrict__ outT)
{
  int lane = threadIdx.x & 31, w = threadIdx.x >> 5;
  int row = blockIdx.x * 8 + w;
  const float* xr = X + (size_t)row * DIMC;
  float x[8], s = 0.f, ss = 0.f;
  #pragma unroll
  for(int i = 0; i < 8; ++i){ x[i] = xr[lane * 8 + i]; s += x[i]; ss += x[i] * x[i]; }
  #pragma unroll
  for(int off = 16; off >= 1; off >>= 1){
    s  += __shfl_xor(s,  off, 32);
    ss += __shfl_xor(ss, off, 32);
  }
  float mean = s * (1.f / 256.f);
  float var  = ss * (1.f / 256.f) - mean * mean;
  float inv  = rsqrtf(var + 1e-5f);
  #pragma unroll
  for(int i = 0; i < 8; ++i){
    int c = lane * 8 + i;
    float v = (x[i] - mean) * inv * g[c] + b[c];
    if(outB) outB[(size_t)row * DIMC + c] = f2bf(v);
    if(outT){                                // final LN writes transposed (b, C, t) fp32
      int bb = row >> 11, t = row & 2047;
      outT[((size_t)bb * DIMC + c) * NTOK + t] = v;
    }
  }
}

// ---------------------------------------------------------------- pos embedding + latitude bias
__global__ __launch_bounds__(256) void k_pos(const float* __restrict__ pw,
    const float* __restrict__ pb, float* __restrict__ tok, float* __restrict__ lat)
{
  const float PI = 3.14159265358979323846f;
  int t = blockIdx.x;
  int i = t >> 6, j = t & 63;
  // bilinear resize of separable analytic grids (256x512 -> 32x64)
  float ch = fminf(fmaxf((i + 0.5f) * 8.f - 0.5f, 0.f), 255.f);
  int i0 = (int)floorf(ch); int i1 = min(i0 + 1, 255); float wr = ch - (float)i0;
  float theta = ((i0 + 0.5f) / 256.f * PI - PI * 0.5f) * (1.f - wr)
              + ((i1 + 0.5f) / 256.f * PI - PI * 0.5f) * wr;
  float cw = fminf(fmaxf((j + 0.5f) * 8.f - 0.5f, 0.f), 511.f);
  int j0 = (int)floorf(cw); int j1 = min(j0 + 1, 511); float wc = cw - (float)j0;
  float phi = ((j0 + 0.5f) / 512.f * 2.f * PI - PI) * (1.f - wc)
            + ((j1 + 0.5f) / 512.f * 2.f * PI - PI) * wc;
  float feats[35];
  #pragma unroll
  for(int k = 0; k < 8; ++k){
    float f = (float)(1 << k);
    feats[4 * k + 0] = sinf(f * phi);   feats[4 * k + 1] = cosf(f * phi);
    feats[4 * k + 2] = sinf(f * theta); feats[4 * k + 3] = cosf(f * theta);
  }
  float ct = cosf(theta);
  float p0 = ct * cosf(phi), p1 = ct * sinf(phi), p2 = sinf(theta);
  float nrm = fmaxf(sqrtf(p0 * p0 + p1 * p1 + p2 * p2), 1e-12f);
  feats[32] = p0 / nrm; feats[33] = p1 / nrm; feats[34] = p2 / nrm;
  int c = threadIdx.x;
  float acc = pb[c];
  #pragma unroll
  for(int f = 0; f < 35; ++f) acc += feats[f] * pw[c * 35 + f];
  tok[(size_t)t * DIMC + c] += acc;
  tok[(size_t)(NTOK + t) * DIMC + c] += acc;
  if(c == 0) lat[t] = logf(fmaxf(fabsf(sinf(theta)), 1e-6f));
}

// ---------------------------------------------------------------- QKV head split (+fold HD^-0.5 into Q)
__global__ void k_qkvsplit(const float* __restrict__ qkv, u16* __restrict__ Q,
                           u16* __restrict__ Kh, u16* __restrict__ Vt){
  int idx = blockIdx.x * blockDim.x + threadIdx.x;
  if(idx >= MROWS * 768) return;
  int r = idx / 768, j = idx - r * 768;
  int which = j >> 8, h = (j >> 5) & 7, d = j & 31;
  int b = r >> 11, t = r & 2047;
  int bh = b * NHEADS + h;
  float v = qkv[idx];
  if(which == 0)      Q [((size_t)bh * NTOK + t) * HDIM + d] = f2bf(v * 0.17677669529663688f);
  else if(which == 1) Kh[((size_t)bh * NTOK + t) * HDIM + d] = f2bf(v);
  else                Vt[((size_t)bh * HDIM + d) * NTOK + t] = f2bf(v);  // V transposed for B-frag
}

// ---------------------------------------------------------------- flash attention (wave = 16 queries of one (b,h))
__global__ __launch_bounds__(256) void k_attn(const u16* __restrict__ Q,
    const u16* __restrict__ Kh, const u16* __restrict__ Vt,
    const float* __restrict__ lat, u16* __restrict__ O)
{
  __shared__ __align__(32) u16 sm[8 * 16 * 32];      // per-wave 16x32 P staging (8KB)
  int lane = threadIdx.x & 31, w = threadIdx.x >> 5;
  int wg = blockIdx.x * 8 + w;                       // 0..2047
  int bh = wg >> 7;                                  // (b*8+h)
  int q0 = (wg & 127) << 4;                          // query tile base
  int half = lane >> 4, ln = lane & 15;
  const u16* qb = Q  + (size_t)bh * NTOK * HDIM;
  const u16* kb = Kh + (size_t)bh * NTOK * HDIM;
  const u16* vb = Vt + (size_t)bh * HDIM * NTOK;
  u16* my = sm + w * 512;

  v16bf aq = load_a_frag(qb, HDIM, q0, 0, lane);     // Q tile, reused for all K tiles
  v8f o0, o1;
  float mrow[8], lrow[8];
  #pragma unroll
  for(int r = 0; r < 8; ++r){ o0[r] = 0.f; o1[r] = 0.f; mrow[r] = -1e30f; lrow[r] = 0.f; }

  for(int j0 = 0; j0 < NTOK; j0 += 32){
    // S = Q*K^T (two 16x16 tiles, K-dim = HD = 32 -> one wmma each)
    v8f z; 
    #pragma unroll
    for(int r = 0; r < 8; ++r) z[r] = 0.f;
    v16bf bk0 = load_b_frag(kb, HDIM, j0,      0, lane);
    v16bf bk1 = load_b_frag(kb, HDIM, j0 + 16, 0, lane);
    v8f s0 = wmma_bf16(aq, bk0, z);
    v8f s1 = wmma_bf16(aq, bk1, z);
    float lat0 = lat[j0 + ln], lat1 = lat[j0 + 16 + ln];
    // online softmax over this 32-column slab
    #pragma unroll
    for(int r = 0; r < 8; ++r){
      s0[r] += lat0; s1[r] += lat1;
      float v = fmaxf(s0[r], s1[r]);
      #pragma unroll
      for(int off = 8; off >= 1; off >>= 1) v = fmaxf(v, __shfl_xor(v, off, 16));
      float nm = fmaxf(mrow[r], v);
      float alpha = __expf(mrow[r] - nm);
      mrow[r] = nm; lrow[r] *= alpha; o0[r] *= alpha; o1[r] *= alpha;
      s0[r] = __expf(s0[r] - nm);
      s1[r] = __expf(s1[r] - nm);
      float rs = s0[r] + s1[r];
      #pragma unroll
      for(int off = 8; off >= 1; off >>= 1) rs += __shfl_xor(rs, off, 16);
      lrow[r] += rs;
      // stage P (bf16) row-major: row = r+8*half, cols ln / 16+ln
      my[(r + 8 * half) * 32 + ln]      = f2bf(s0[r]);
      my[(r + 8 * half) * 32 + 16 + ln] = f2bf(s1[r]);
    }
    __syncthreads();
    // P (16x32) as A fragment from LDS
    const u16* pp = my + ln * 32 + 8 * half;
    v16bf ap = cat8(*(const v8bf*)pp, *(const v8bf*)(pp + 16));
    // O += P * V  (V^T row-major [d][j], so columns d load K-contiguous)
    v16bf bv0 = load_b_frag(vb, NTOK, 0,  j0, lane);
    v16bf bv1 = load_b_frag(vb, NTOK, 16, j0, lane);
    o0 = wmma_bf16(ap, bv0, o0);
    o1 = wmma_bf16(ap, bv1, o1);
    __syncthreads();
  }
  // write O as (b, t, h*32+d) bf16 for the proj GEMM
  int b_ = bh >> 3, h = bh & 7;
  #pragma unroll
  for(int r = 0; r < 8; ++r){
    float inv = 1.f / lrow[r];
    int row = q0 + r + 8 * half;
    size_t base = ((size_t)b_ * NTOK + row) * DIMC + h * HDIM;
    O[base + ln]      = f2bf(o0[r] * inv);
    O[base + 16 + ln] = f2bf(o1[r] * inv);
  }
}

// ---------------------------------------------------------------- host orchestration
extern "C" void kernel_launch(void* const* d_in, const int* in_sizes, int n_in,
                              void* d_out, int out_size, void* d_ws, size_t ws_size,
                              hipStream_t stream)
{
  (void)in_sizes; (void)n_in; (void)out_size; (void)ws_size;
  const float* x       = (const float*)d_in[0];
  const float* conv1_w = (const float*)d_in[1];
  const float* bn1_g   = (const float*)d_in[2];
  const float* bn1_b   = (const float*)d_in[3];
  const float* bn1_m   = (const float*)d_in[4];
  const float* bn1_v   = (const float*)d_in[5];
  const float* conv2_w = (const float*)d_in[6];
  const float* bn2_g   = (const float*)d_in[7];
  const float* bn2_b   = (const float*)d_in[8];
  const float* bn2_m   = (const float*)d_in[9];
  const float* bn2_v   = (const float*)d_in[10];
  const float* pos_w   = (const float*)d_in[11];
  const float* pos_b   = (const float*)d_in[12];
  const float* ln1_g   = (const float*)d_in[13];
  const float* ln1_b   = (const float*)d_in[14];
  const float* qkv_w   = (const float*)d_in[15];
  const float* qkv_b   = (const float*)d_in[16];
  const float* proj_w  = (const float*)d_in[17];
  const float* proj_b  = (const float*)d_in[18];
  const float* ln2_g   = (const float*)d_in[19];
  const float* ln2_b   = (const float*)d_in[20];
  const float* fc1_w   = (const float*)d_in[21];
  const float* fc1_b   = (const float*)d_in[22];
  const float* fc2_w   = (const float*)d_in[23];
  const float* fc2_b   = (const float*)d_in[24];
  const float* norm_g  = (const float*)d_in[25];
  const float* norm_b  = (const float*)d_in[26];

  char* ws = (char*)d_ws;
  size_t off = 0;
  auto alloc = [&](size_t bytes) -> void* {
    off = (off + 255) & ~(size_t)255;
    void* p = ws + off; off += bytes; return p;
  };
  // bf16 weights
  u16* W1    = (u16*)alloc((size_t)128 * 192 * 2);
  u16* W2    = (u16*)alloc((size_t)256 * 1152 * 2);
  u16* Wqkv  = (u16*)alloc((size_t)8 * 768 * 256 * 2);
  u16* Wproj = (u16*)alloc((size_t)8 * 256 * 256 * 2);
  u16* Wfc1  = (u16*)alloc((size_t)8 * 1024 * 256 * 2);
  u16* Wfc2  = (u16*)alloc((size_t)8 * 256 * 1024 * 2);
  float* bn1s = (float*)alloc(128 * 4);
  float* bn1t = (float*)alloc(128 * 4);
  float* bn2s = (float*)alloc(256 * 4);
  float* bn2t = (float*)alloc(256 * 4);
  // activations
  u16*   A1   = (u16*)alloc((size_t)MROWS * 192 * 2);
  u16*   y1b  = (u16*)alloc((size_t)MROWS * 128 * 2);
  u16*   A2   = (u16*)alloc((size_t)MROWS * 1152 * 2);
  float* tok  = (float*)alloc((size_t)MROWS * DIMC * 4);
  float* lat  = (float*)alloc((size_t)NTOK * 4);
  u16*   xln  = (u16*)alloc((size_t)MROWS * DIMC * 2);
  float* qkvb = (float*)alloc((size_t)MROWS * 768 * 4);
  u16*   Qh   = (u16*)alloc((size_t)16 * NTOK * HDIM * 2);
  u16*   Khb  = (u16*)alloc((size_t)16 * NTOK * HDIM * 2);
  u16*   Vtb  = (u16*)alloc((size_t)16 * HDIM * NTOK * 2);
  u16*   Obuf = (u16*)alloc((size_t)MROWS * DIMC * 2);
  u16*   h1   = (u16*)alloc((size_t)MROWS * HIDDENC * 2);

  auto cvt = [&](const float* s, u16* d, int n){
    k_f2bf<<<(n + 255) / 256, 256, 0, stream>>>(s, d, n);
  };
  auto gemm = [&](const u16* A, const u16* W, int K, int N,
                  const float* scale, const float* shift, int act, int mode,
                  float* outF, u16* outB){
    dim3 g(MROWS / 128, N / 64);
    k_gemm<<<g, 256, 0, stream>>>(A, W, K, N, scale, shift, act, mode, outF, outB);
  };

  // ---- weight prep
  cvt(conv1_w, W1, 128 * 192);
  cvt(conv2_w, W2, 256 * 1152);
  cvt(qkv_w,  Wqkv, 8 * 768 * 256);
  cvt(proj_w, Wproj, 8 * 256 * 256);
  cvt(fc1_w,  Wfc1, 8 * 1024 * 256);
  cvt(fc2_w,  Wfc2, 8 * 256 * 1024);
  k_bnfuse<<<1, 128, 0, stream>>>(bn1_g, bn1_b, bn1_m, bn1_v, bn1s, bn1t, 128);
  k_bnfuse<<<1, 256, 0, stream>>>(bn2_g, bn2_b, bn2_m, bn2_v, bn2s, bn2t, 256);

  // ---- patchify conv1 -> BN+GELU (bf16)
  k_im2col1<<<(MROWS * 192 + 255) / 256, 256, 0, stream>>>(x, A1);
  gemm(A1, W1, 192, 128, bn1s, bn1t, /*gelu*/1, /*bf16*/4, nullptr, y1b);

  // ---- conv2 (3x3) -> BN+GELU -> tokens fp32
  k_im2col2<<<(MROWS * 1152 + 255) / 256, 256, 0, stream>>>(y1b, A2);
  gemm(A2, W2, 1152, 256, bn2s, bn2t, 1, /*f32 store*/1, tok, nullptr);

  // ---- positional embedding + latitude bias
  k_pos<<<NTOK, 256, 0, stream>>>(pos_w, pos_b, tok, lat);

  // ---- transformer layers
  for(int L = 0; L < 8; ++L){
    k_ln<<<MROWS / 8, 256, 0, stream>>>(tok, ln1_g + L * 256, ln1_b + L * 256, xln, nullptr);
    gemm(xln, Wqkv + (size_t)L * 768 * 256, 256, 768, nullptr, qkv_b + L * 768, 0, 1, qkvb, nullptr);
    k_qkvsplit<<<(MROWS * 768 + 255) / 256, 256, 0, stream>>>(qkvb, Qh, Khb, Vtb);
    k_attn<<<256, 256, 0, stream>>>(Qh, Khb, Vtb, lat, Obuf);
    gemm(Obuf, Wproj + (size_t)L * 256 * 256, 256, 256, nullptr, proj_b + L * 256, 0, /*+=*/2, tok, nullptr);
    k_ln<<<MROWS / 8, 256, 0, stream>>>(tok, ln2_g + L * 256, ln2_b + L * 256, xln, nullptr);
    gemm(xln, Wfc1 + (size_t)L * 1024 * 256, 256, 1024, nullptr, fc1_b + L * 1024, 1, 4, nullptr, h1);
    gemm(h1, Wfc2 + (size_t)L * 256 * 1024, 1024, 256, nullptr, fc2_b + L * 256, 0, 2, tok, nullptr);
  }

  // ---- final LN, transposed fp32 output (b, C, h, w)
  k_ln<<<MROWS / 8, 256, 0, stream>>>(tok, norm_g, norm_b, nullptr, (float*)d_out);
}
